// SparseUnet_52158082843101
// MI455X (gfx1250) — matrix operand order repaired and checked by reading, not compiled, not run on hardware
//
#include <hip/hip_runtime.h>
#include <hip/hip_bf16.h>

typedef __attribute__((ext_vector_type(16))) __bf16 v16bf;
typedef __attribute__((ext_vector_type(8)))  float  v8f;
typedef __bf16 bf;

#define BN_EPS 1e-3f
#define IDX1D  long long i = (long long)blockIdx.x * blockDim.x + threadIdx.x

// ---------------------------------------------------------------------------
// Implicit-GEMM conv (normal / strided / lhs-dilated) with bf16 WMMA.
// Feature layout: [D][H][W][C] channels-last bf16.  Weight: [Cout][Kpad] bf16,
// k = tap*Cin + ci, tap = (fd*KH+fh)*KW+fw, Kpad = ceil(K,64) zero-padded.
// Per-row tap->voxel tables are precomputed in LDS so the K-loop has no
// integer div/mod or bounds math.  Global fetch of chunk kb+1 is software-
// pipelined in registers while WMMAs consume chunk kb from LDS.
// ---------------------------------------------------------------------------
struct ConvP {
  const bf* X; const bf* W; bf* Y;
  int Din, Hin, Win, Do, Ho, Wo;
  int Cin, cinShift, Cout;
  int KH_KW, KW, ntaps;
  int padD, padH, padW;
  int stride, ldil;
  int Kpad;
};

__global__ __launch_bounds__(256) void k_conv_wmma(ConvP p, int Ntile) {
  __shared__ __align__(16) bf lA[64 * 72];   // 64 rows x 64 k, stride 72 halves
  __shared__ __align__(16) bf lB[64 * 72];   // 64 cols x 64 k, stride 72 halves
  __shared__ int tab[64 * 32];               // per-row tap -> input voxel (-1 oob)
  const int t = threadIdx.x;
  const int wid = t >> 5, lane = t & 31;
  const int m0 = blockIdx.x * 64;
  const int n0 = blockIdx.y * Ntile;
  const int rowA = t >> 2, seg = t & 3;

  // ---- init tap table to -1 (covers Kpad padding taps too)
#pragma unroll
  for (int j = 0; j < 8; ++j) tab[t + j * 256] = -1;
  __syncthreads();

  // ---- once-per-thread output coord decomposition + tap table build
  {
    const int rg = m0 + rowA;
    const int od = rg / (p.Ho * p.Wo);
    const int rr = rg % (p.Ho * p.Wo);
    const int oh = rr / p.Wo, ow = rr % p.Wo;
    const int bd = od * p.stride - p.padD;
    const int bh = oh * p.stride - p.padH;
    const int bw = ow * p.stride - p.padW;
    for (int tap = seg; tap < p.ntaps; tap += 4) {
      const int fd = tap / p.KH_KW, tr = tap % p.KH_KW;
      const int fh = tr / p.KW, fw = tr % p.KW;
      const int td = bd + fd, th = bh + fh, tw = bw + fw;
      int id = td, ih = th, iw = tw;
      bool ok = true;
      if (p.ldil == 2) {                      // transposed conv: even taps only
        ok = (((td | th | tw) & 1) == 0);
        id = td >> 1; ih = th >> 1; iw = tw >> 1;
      }
      ok = ok && id >= 0 && id < p.Din && ih >= 0 && ih < p.Hin &&
           iw >= 0 && iw < p.Win;
      if (ok) tab[rowA * 32 + tap] = (id * p.Hin + ih) * p.Win + iw;
    }
  }
  __syncthreads();

  const int kk0 = seg * 8;          // two 16B segments per thread: kk0, kk0+32
  const int cmask = p.Cin - 1;
  const int nBig = p.Kpad >> 6;
  const int* tabRow = &tab[rowA * 32];
  const int mi = wid & 3, nsb = wid >> 2;
  const int nAcc = Ntile >> 5;
  const int lrow = lane & 15, lhalf = lane >> 4;
  const bf* wBase = p.W + (long long)(n0 + rowA) * p.Kpad + kk0; // used if rowA<Ntile

  uint4 avr0, avr1, bvr0, bvr1;
  avr0.x = avr0.y = avr0.z = avr0.w = 0u;
  avr1 = avr0; bvr0 = avr0; bvr1 = avr0;

  auto fetch = [&](int kb) {
#pragma unroll
    for (int h = 0; h < 2; ++h) {
      const int k0 = (kb << 6) + kk0 + h * 32;
      uint4 av; av.x = av.y = av.z = av.w = 0u;
      const int vox = tabRow[k0 >> p.cinShift];
      if (vox >= 0)
        av = *(const uint4*)(p.X + ((long long)vox << p.cinShift) + (k0 & cmask));
      if (h == 0) avr0 = av; else avr1 = av;
    }
    if (rowA < Ntile) {
      const bf* wp = wBase + (kb << 6);
      bvr0 = *(const uint4*)wp;
      bvr1 = *(const uint4*)(wp + 32);
      if (kb + 2 < nBig) __builtin_prefetch(wp + 128, 0, 1);  // global_prefetch
    }
  };

  v8f acc0 = {}; v8f acc1 = {};
  fetch(0);

  for (int kb = 0; kb < nBig; ++kb) {
    // ---- publish staged chunk kb to LDS
    *(uint4*)(&lA[rowA * 72 + kk0])      = avr0;
    *(uint4*)(&lA[rowA * 72 + kk0 + 32]) = avr1;
    if (rowA < Ntile) {
      *(uint4*)(&lB[rowA * 72 + kk0])      = bvr0;
      *(uint4*)(&lB[rowA * 72 + kk0 + 32]) = bvr1;
    }
    __syncthreads();
    // ---- start global fetch of chunk kb+1 (overlaps with WMMAs below)
    if (kb + 1 < nBig) fetch(kb + 1);

    // ---- 2 x (K=32) WMMA per accumulator per stage
#pragma unroll
    for (int c2 = 0; c2 < 2; ++c2) {
      v16bf a;
      {
        const bf* ap = &lA[(mi * 16 + lrow) * 72 + c2 * 32 + lhalf * 16];
        ((uint4*)&a)[0] = *(const uint4*)ap;        // 16 consecutive K
        ((uint4*)&a)[1] = *(const uint4*)(ap + 8);
      }
      {
        v16bf b;
        const bf* bp = &lB[(nsb * 16 + lrow) * 72 + c2 * 32 + lhalf * 16];
        ((uint4*)&b)[0] = *(const uint4*)bp;        // 16 consecutive K
        ((uint4*)&b)[1] = *(const uint4*)(bp + 8);
        acc0 = __builtin_amdgcn_wmma_f32_16x16x32_bf16(false, a, false, b,
                                                       (short)0, acc0, false, false);
      }
      if (nAcc == 2) {
        v16bf b;
        const bf* bp = &lB[((nsb + 2) * 16 + lrow) * 72 + c2 * 32 + lhalf * 16];
        ((uint4*)&b)[0] = *(const uint4*)bp;
        ((uint4*)&b)[1] = *(const uint4*)(bp + 8);
        acc1 = __builtin_amdgcn_wmma_f32_16x16x32_bf16(false, a, false, b,
                                                       (short)0, acc1, false, false);
      }
    }
    __syncthreads();
  }

  // ---- epilogue: C/D layout (vgpr r -> M=r(+8), lane%16 -> N)
#pragma unroll
  for (int r = 0; r < 8; ++r) {
    const long long m = m0 + mi * 16 + r + 8 * lhalf;
    p.Y[m * p.Cout + (n0 + nsb * 16 + lrow)] = (bf)acc0[r];
  }
  if (nAcc == 2) {
#pragma unroll
    for (int r = 0; r < 8; ++r) {
      const long long m = m0 + mi * 16 + r + 8 * lhalf;
      p.Y[m * p.Cout + (n0 + (nsb + 2) * 16 + lrow)] = (bf)acc1[r];
    }
  }
}

// ---------------------------------------------------------------------------
// Support kernels
// ---------------------------------------------------------------------------
__global__ void k_wconv(const float* w, bf* o, int Cout, int Cin,
                        int KD, int KH, int KW, int Kpad) {
  IDX1D;
  const long long n = (long long)Cout * Kpad;
  if (i >= n) return;
  const int co = (int)(i / Kpad), k = (int)(i % Kpad);
  const int K = Cin * KD * KH * KW;
  float v = 0.f;
  if (k < K) {
    const int tap = k / Cin, ci = k % Cin;
    const int fd = tap / (KH * KW), tr = tap % (KH * KW);
    const int fh = tr / KW, fw = tr % KW;
    v = w[(((long long)(co * Cin + ci) * KD + fd) * KH + fh) * KW + fw];
  }
  o[i] = (bf)v;
}

__global__ void k_scatter(const float* vf, const int* vc, float* dense32,
                          float* occ, int Nv) {
  IDX1D;
  if (i >= (long long)Nv * 16) return;
  const int n = (int)(i >> 4), c = (int)(i & 15);
  const int z = vc[n * 4 + 1], y = vc[n * 4 + 2], x = vc[n * 4 + 3];
  const long long v = ((long long)z * 96 + y) * 96 + x;
  atomicAdd(&dense32[v * 16 + c], vf[(long long)n * 16 + c]);
  if (c == 0) atomicAdd(&occ[v], 1.0f);
}

__global__ void k_densify(const float* dense32, const float* occ, bf* dense16,
                          bf* m, float* cnt, long long S) {
  IDX1D;
  if (i >= S) return;
  const float o = occ[i];
  m[i] = (o > 0.f) ? (bf)1.0f : (bf)0.0f;
  if (o > 0.f) atomicAdd(cnt, 1.0f);
#pragma unroll
  for (int c = 0; c < 16; ++c) dense16[i * 16 + c] = (bf)dense32[i * 16 + c];
}

__global__ void k_maskdown(const bf* mi_, bf* mo, float* cnt,
                           int Din, int Hin, int Win, int Do, int Ho, int Wo,
                           int pd, int ph, int pw) {
  IDX1D;
  const long long S = (long long)Do * Ho * Wo;
  if (i >= S) return;
  const int od = (int)(i / (Ho * Wo)), rr = (int)(i % (Ho * Wo));
  const int oh = rr / Wo, ow = rr % Wo;
  float act = 0.f;
  for (int fd = 0; fd < 3; ++fd) {
    const int id = od * 2 + fd - pd; if (id < 0 || id >= Din) continue;
    for (int fh = 0; fh < 3; ++fh) {
      const int ih = oh * 2 + fh - ph; if (ih < 0 || ih >= Hin) continue;
      for (int fw = 0; fw < 3; ++fw) {
        const int iw = ow * 2 + fw - pw; if (iw < 0 || iw >= Win) continue;
        if ((float)mi_[((long long)id * Hin + ih) * Win + iw] > 0.f) act = 1.f;
      }
    }
  }
  mo[i] = (bf)act;
  if (act > 0.f) atomicAdd(cnt, 1.f);
}

__global__ void k_bn_stats(const bf* X, const bf* msk, float* sums, float* sqs,
                           int C, int log2C, long long S) {
  __shared__ float sS[256], sQ[256];
  const int t = threadIdx.x;
  sS[t] = 0.f; sQ[t] = 0.f;
  __syncthreads();
  const int c = t & (C - 1);
  const int sub = t >> log2C;
  const int vpb = 256 >> log2C;
  float s = 0.f, q = 0.f;
  for (long long v = (long long)blockIdx.x * vpb + sub; v < S;
       v += (long long)gridDim.x * vpb) {
    if ((float)msk[v] > 0.f) {
      const float x = (float)X[(v << log2C) + c];
      s += x; q += x * x;
    }
  }
  atomicAdd(&sS[c], s); atomicAdd(&sQ[c], q);
  __syncthreads();
  if (t < C) { atomicAdd(&sums[t], sS[t]); atomicAdd(&sqs[t], sQ[t]); }
}

__global__ void k_bn_fin(const float* sums, const float* sqs, const float* g,
                         const float* b, const float* cnt, float* scale,
                         float* shift, int C) {
  const int c = threadIdx.x;
  if (c >= C) return;
  float n = cnt[0]; if (n < 1.f) n = 1.f;
  const float mean = sums[c] / n;
  float var = sqs[c] / n - mean * mean; if (var < 0.f) var = 0.f;
  const float sc = g[c] * rsqrtf(var + BN_EPS);
  scale[c] = sc;
  shift[c] = b[c] - mean * sc;
}

__global__ void k_bn_apply(const bf* X, const bf* msk, const float* scale,
                           const float* shift, bf* Y, int log2C, long long n,
                           int relu) {
  IDX1D;
  if (i >= n) return;
  const int c = (int)(i & ((1 << log2C) - 1));
  const long long v = i >> log2C;
  float y = (float)X[i] * scale[c] + shift[c];
  if (relu && y < 0.f) y = 0.f;
  if (!((float)msk[v] > 0.f)) y = 0.f;
  Y[i] = (bf)y;
}

__global__ void k_add(const bf* a, const bf* b, bf* y, long long n) {
  IDX1D; if (i >= n) return;
  y[i] = (bf)((float)a[i] + (float)b[i]);
}

__global__ void k_resid(const bf* a, const bf* lat, bf* y, long long n) {
  IDX1D; if (i >= n) return;
  float v = (float)a[i] + (float)lat[i];
  y[i] = (bf)(v > 0.f ? v : 0.f);
}

__global__ void k_catcopy(const bf* src, bf* dst, int log2C, int off, long long n) {
  IDX1D; if (i >= n) return;
  const int C = 1 << log2C;
  const int c = (int)(i & (C - 1));
  const long long v = i >> log2C;
  dst[v * (2 * C) + off + c] = src[i];
}

__global__ void k_red(const bf* xm, const bf* cat, bf* y, int log2C, long long n) {
  IDX1D; if (i >= n) return;
  const int c = (int)(i & ((1 << log2C) - 1));
  const long long v = i >> log2C;
  const bf* row = cat + (v << (log2C + 1));
  y[i] = (bf)((float)xm[i] + (float)row[2 * c] + (float)row[2 * c + 1]);
}

__global__ void k_gather(const bf* u, const int* vc, float* out, int Nv) {
  IDX1D;
  if (i >= (long long)Nv * 32) return;
  const int n = (int)(i >> 5), c = (int)(i & 31);
  const int z = vc[n * 4 + 1], y = vc[n * 4 + 2], x = vc[n * 4 + 3];
  const long long v = ((long long)z * 96 + y) * 96 + x;
  out[i] = (float)u[v * 32 + c];
}

// ---------------------------------------------------------------------------
// Host orchestration
// ---------------------------------------------------------------------------
static inline int ilog2i(int x) { int r = 0; while ((1 << r) < x) ++r; return r; }

extern "C" void kernel_launch(void* const* d_in, const int* in_sizes, int n_in,
                              void* d_out, int out_size, void* d_ws, size_t ws_size,
                              hipStream_t stream) {
  (void)n_in; (void)out_size; (void)ws_size;
  const float* vf = (const float*)d_in[0];
  const int*   vc = (const int*)d_in[1];
  const int Nv = in_sizes[0] / 16;

  // param leaves: alphabetical pytree order, (b,g,w) per layer, starting d_in[3]
  // 0 c1, 1 c2_0, 2 c2_1, 3 c2_2, 4 c3_0, 5 c3_1, 6 c3_2, 7 c4_0, 8 c4_1,
  // 9 c4_2, 10 c5, 11 ci_c1, 12 ci_c12, 13 ci_c2, 14 ci_c3, 15 inv2, 16 inv3,
  // 17 inv4, 18 m1, 19 m2, 20 m3, 21 m4, 22 t1_1, 23 t1_2, 24 t2_1, 25 t2_2,
  // 26 t3_1, 27 t3_2, 28 t4_1, 29 t4_2
  static const int LC[30][5] = {
    {32,32,3,3,3},{64,32,3,3,3},{64,64,3,3,3},{64,64,3,3,3},
    {128,64,3,3,3},{128,128,3,3,3},{128,128,3,3,3},{128,128,3,3,3},
    {128,128,3,3,3},{128,128,3,3,3},{32,32,3,3,3},
    {32,16,1,3,3},{32,32,3,1,3},{32,16,3,1,3},{32,32,1,3,3},
    {32,64,3,3,3},{64,128,3,3,3},{128,128,3,3,3},
    {32,64,3,3,3},{64,128,3,3,3},{128,256,3,3,3},{128,256,3,3,3},
    {32,32,3,3,3},{32,32,3,3,3},{64,64,3,3,3},{64,64,3,3,3},
    {128,128,3,3,3},{128,128,3,3,3},{128,128,3,3,3},{128,128,3,3,3}};

  static const int DD[4] = {33, 17, 9, 4};
  static const int HH[4] = {96, 48, 24, 12};
  static const int WWD[4] = {96, 48, 24, 12};
  long long S[4];
  for (int l = 0; l < 4; ++l) S[l] = (long long)DD[l] * HH[l] * WWD[l];

  char* basep = (char*)d_ws; size_t off = 0;
  auto alloc = [&](size_t bytes) -> char* {
    off = (off + 255) & ~(size_t)255;
    char* p = basep + off; off += bytes; return p;
  };

  float* dense32 = (float*)alloc(S[0] * 16 * 4);
  float* occ     = (float*)alloc(S[0] * 4);
  float* counts  = (float*)alloc(4 * sizeof(float));
  float* sums    = (float*)alloc(256 * 4);
  float* sqs     = (float*)alloc(256 * 4);
  float* scl     = (float*)alloc(256 * 4);
  float* shf     = (float*)alloc(256 * 4);
  bf* dense16 = (bf*)alloc(S[0] * 16 * 2);
  bf* mk[4]; for (int l = 0; l < 4; ++l) mk[l] = (bf*)alloc(S[l] * 2);
  bf* x0 = (bf*)alloc(S[0] * 32 * 2);
  bf* x1 = (bf*)alloc(S[0] * 32 * 2);
  bf* x2 = (bf*)alloc(S[1] * 64 * 2);
  bf* x3 = (bf*)alloc(S[2] * 128 * 2);
  bf* x4 = (bf*)alloc(S[3] * 128 * 2);
  bf* u4 = (bf*)alloc(S[2] * 128 * 2);
  bf* u3 = (bf*)alloc(S[1] * 64 * 2);
  bf* u2 = (bf*)alloc(S[0] * 32 * 2);
  bf* tA = (bf*)alloc(S[0] * 64 * 2);   // concat buffers (max 2C at level 0)
  bf* tB = (bf*)alloc(S[0] * 32 * 2);   // raw conv output scratch
  bf* tC = (bf*)alloc(S[0] * 32 * 2);
  bf* tD = (bf*)alloc(S[0] * 32 * 2);

  // weight conversion (fp32 OIDHW -> bf16 [Cout][Kpad], Kpad = ceil(K,64))
  bf* Wt[30]; int Kpad[30];
  for (int li = 0; li < 30; ++li) {
    const int Cout = LC[li][0], Cin = LC[li][1];
    const int K = Cin * LC[li][2] * LC[li][3] * LC[li][4];
    Kpad[li] = (K + 63) & ~63;
    Wt[li] = (bf*)alloc((size_t)Cout * Kpad[li] * 2);
    const long long n = (long long)Cout * Kpad[li];
    k_wconv<<<dim3((unsigned)((n + 255) / 256)), dim3(256), 0, stream>>>(
        (const float*)d_in[3 + 3 * li + 2], Wt[li], Cout, Cin,
        LC[li][2], LC[li][3], LC[li][4], Kpad[li]);
  }

  (void)hipMemsetAsync(dense32, 0, S[0] * 16 * 4, stream);
  (void)hipMemsetAsync(occ, 0, S[0] * 4, stream);
  (void)hipMemsetAsync(counts, 0, 16, stream);
  {
    const long long n = (long long)Nv * 16;
    k_scatter<<<dim3((unsigned)((n + 255) / 256)), dim3(256), 0, stream>>>(
        vf, vc, dense32, occ, Nv);
  }
  k_densify<<<dim3((unsigned)((S[0] + 255) / 256)), dim3(256), 0, stream>>>(
      dense32, occ, dense16, mk[0], &counts[0], S[0]);

  auto ew = [](long long n) { return dim3((unsigned)((n + 255) / 256)); };

  auto conv = [&](int li, const bf* X, bf* Y, int lin, int lout,
                  int pd, int ph, int pw_, int stride, int ldil) {
    ConvP p;
    p.X = X; p.W = Wt[li]; p.Y = Y;
    p.Din = DD[lin]; p.Hin = HH[lin]; p.Win = WWD[lin];
    p.Do = DD[lout]; p.Ho = HH[lout]; p.Wo = WWD[lout];
    p.Cin = LC[li][1]; p.cinShift = ilog2i(p.Cin); p.Cout = LC[li][0];
    p.KW = LC[li][4]; p.KH_KW = LC[li][3] * LC[li][4];
    p.ntaps = LC[li][2] * p.KH_KW;
    p.padD = pd; p.padH = ph; p.padW = pw_;
    p.stride = stride; p.ldil = ldil; p.Kpad = Kpad[li];
    const long long M = (long long)p.Do * p.Ho * p.Wo;
    const int Ntile = (p.Cout >= 64) ? 64 : 32;
    dim3 grid((unsigned)(M / 64), (unsigned)(p.Cout / Ntile));
    k_conv_wmma<<<grid, dim3(256), 0, stream>>>(p, Ntile);
  };

  auto bn = [&](int li, const bf* raw, int lev, bf* Y, bool relu) {
    const int C = LC[li][0], lc = ilog2i(C);
    const long long n = S[lev] * C;
    (void)hipMemsetAsync(sums, 0, C * 4, stream);
    (void)hipMemsetAsync(sqs, 0, C * 4, stream);
    k_bn_stats<<<dim3(128), dim3(256), 0, stream>>>(raw, mk[lev], sums, sqs, C,
                                                    lc, S[lev]);
    k_bn_fin<<<dim3(1), dim3(256), 0, stream>>>(
        sums, sqs, (const float*)d_in[3 + 3 * li + 1],
        (const float*)d_in[3 + 3 * li + 0], &counts[lev], scl, shf, C);
    k_bn_apply<<<ew(n), dim3(256), 0, stream>>>(raw, mk[lev], scl, shf, Y, lc,
                                                n, relu ? 1 : 0);
  };

  auto subm = [&](int li, const bf* X, bf* Y, int lev, bool relu) {
    conv(li, X, tB, lev, lev, (LC[li][2] - 1) / 2, (LC[li][3] - 1) / 2,
         (LC[li][4] - 1) / 2, 1, 1);
    bn(li, tB, lev, Y, relu);
  };
  auto down = [&](int li, const bf* X, bf* Y, int lev, int pd, int ph, int pw_) {
    conv(li, X, tB, lev, lev + 1, pd, ph, pw_, 2, 1);
    k_maskdown<<<ew(S[lev + 1]), dim3(256), 0, stream>>>(
        mk[lev], mk[lev + 1], &counts[lev + 1], DD[lev], HH[lev], WWD[lev],
        DD[lev + 1], HH[lev + 1], WWD[lev + 1], pd, ph, pw_);
    bn(li, tB, lev + 1, Y, true);
  };
  auto up = [&](int li, const bf* X, bf* Y, int lin, int lout,
                int pd, int ph, int pw_) {
    conv(li, X, tB, lin, lout, pd, ph, pw_, 1, 2);
    bn(li, tB, lout, Y, true);
  };
  auto ur = [&](int lt1, int lt2, int lm, const bf* xlat, const bf* xbot,
                int lev) {
    const int C = LC[lt1][0], lc = ilog2i(C);
    const long long n = S[lev] * C;
    subm(lt1, xlat, tC, lev, true);
    subm(lt2, tC, tD, lev, false);
    k_resid<<<ew(n), dim3(256), 0, stream>>>(tD, xlat, tC, n);     // xt
    k_catcopy<<<ew(n), dim3(256), 0, stream>>>(xbot, tA, lc, 0, n);
    k_catcopy<<<ew(n), dim3(256), 0, stream>>>(tC, tA, lc, C, n);
    conv(lm, tA, tB, lev, lev, 1, 1, 1, 1, 1);                     // conv_m
    bn(lm, tB, lev, tD, true);                                     // xm
    k_red<<<ew(n), dim3(256), 0, stream>>>(tD, tA, tC, lc, n);     // xm + red
  };

  // ResContextBlock
  subm(11, dense16, tC, 0, true);
  subm(12, tC, tC, 0, true);
  subm(13, dense16, tD, 0, true);
  subm(14, tD, tD, 0, true);
  { const long long n = S[0] * 32;
    k_add<<<ew(n), dim3(256), 0, stream>>>(tC, tD, x0, n); }
  // encoder
  subm(0, x0, x1, 0, true);
  down(1, x1, x2, 0, 1, 1, 1);
  subm(2, x2, x2, 1, true);
  subm(3, x2, x2, 1, true);
  down(4, x2, x3, 1, 1, 1, 1);
  subm(5, x3, x3, 2, true);
  subm(6, x3, x3, 2, true);
  down(7, x3, x4, 2, 0, 1, 1);
  subm(8, x4, x4, 3, true);
  subm(9, x4, x4, 3, true);
  // decoder (UR blocks + inverse convs; lo-pads from _up_pads)
  ur(28, 29, 21, x4, x4, 3);
  up(17, tC, u4, 3, 2, 2, 1, 1);
  ur(26, 27, 20, x3, u4, 2);
  up(16, tC, u3, 2, 1, 1, 1, 1);
  ur(24, 25, 19, x2, u3, 1);
  up(15, tC, u2, 1, 0, 1, 1, 1);
  ur(22, 23, 18, x1, u2, 0);
  subm(10, tC, tD, 0, true);
  { const long long n = (long long)Nv * 32;
    k_gather<<<ew(n), dim3(256), 0, stream>>>(tD, vc, (float*)d_out, Nv); }
}
